// SmolLM3Attention_33895881900380
// MI455X (gfx1250) — matrix operand order repaired and checked by reading, not compiled, not run on hardware
//
#include <hip/hip_runtime.h>
#include <math.h>

// ---------------------------------------------------------------------------
// Types for CDNA5 WMMA (gfx1250, wave32)
// ---------------------------------------------------------------------------
typedef __attribute__((ext_vector_type(16))) __bf16          v16bf;
typedef __attribute__((ext_vector_type(8)))  float           v8f;
typedef __attribute__((ext_vector_type(8)))  unsigned short  v8us;

union BF16Frag {
    v16bf bf;
    v8us  us[2];
};

__device__ __forceinline__ unsigned short f2bf(float f) {
    unsigned u = __float_as_uint(f);
    u += 0x7FFFu + ((u >> 16) & 1u);       // round-to-nearest-even
    return (unsigned short)(u >> 16);
}

__device__ __forceinline__ v8f wmma_bf16(const BF16Frag& a, const BF16Frag& b, v8f c) {
    return __builtin_amdgcn_wmma_f32_16x16x32_bf16(
        /*neg_a=*/false, a.bf, /*neg_b=*/false, b.bf,
        /*c_mod=*/(short)0, c, /*reuse_a=*/false, /*reuse_b=*/false);
}

// Problem constants (from reference)
#define BATCH 2
#define SLEN  2048
#define DMODEL 2048
#define NHEAD 16
#define NKV   4
#define HDIM  128
#define NTOK  (BATCH * SLEN)   // 4096

// ---------------------------------------------------------------------------
// 1) fp32 -> bf16 elementwise convert
// ---------------------------------------------------------------------------
__global__ void f2bf_kernel(const float* __restrict__ x,
                            unsigned short* __restrict__ y, size_t n) {
    size_t i = (size_t)blockIdx.x * blockDim.x + threadIdx.x;
    if (i < n) y[i] = f2bf(x[i]);
}

// ---------------------------------------------------------------------------
// 2) Pack W (KxN, f32 row-major) into WMMA B-fragment-major bf16:
//    layout: [nt][kt][lane 0..31][16 bf16]; lane L covers col n=L%16,
//    K-half = L/16, 16 contiguous K values.
// ---------------------------------------------------------------------------
__global__ void packw_kernel(const float* __restrict__ W,
                             unsigned short* __restrict__ P, int K, int N) {
    int tid   = blockIdx.x * blockDim.x + threadIdx.x;
    int Kt    = K >> 5;
    int total = Kt * (N >> 4) * 32;
    if (tid >= total) return;
    int lane  = tid & 31;
    int frag  = tid >> 5;
    int kt    = frag % Kt;
    int nt    = frag / Kt;
    int n     = lane & 15;
    int halfk = lane >> 4;
    unsigned short* dst = P + (size_t)tid * 16;
#pragma unroll
    for (int j = 0; j < 16; ++j) {
        float v = W[(size_t)(kt * 32 + halfk * 16 + j) * N + nt * 16 + n];
        dst[j] = f2bf(v);
    }
}

// ---------------------------------------------------------------------------
// 3) GEMM: C(MxN, f32) = A(MxK bf16 row-major) * Bpacked(KxN bf16 frag-major)
//    One wave computes a 32x64 C tile: 2 A-frags x 4 B-frags = 8 WMMAs per
//    32-K step. Manually unrolled by 2 with two NAMED buffer sets so all
//    fragment indexing is compile-time constant (no scratch spills), and
//    pointer-advance addressing so loads use immediate offsets.
//    Requires Kt = K/32 even (true for all calls here: Kt = 64).
// ---------------------------------------------------------------------------
__global__ __launch_bounds__(256)
void gemm_bf16_kernel(const unsigned short* __restrict__ A,
                      const unsigned short* __restrict__ Bp,
                      float* __restrict__ C, int M, int N, int K) {
    const int lane   = threadIdx.x & 31;
    const int wid    = blockIdx.x * (blockDim.x >> 5) + (threadIdx.x >> 5);
    const int ntiles = N >> 6;                 // 64-wide wave tiles
    const int mt     = wid / ntiles;           // 32-row tile index
    const int nt0    = (wid % ntiles) << 2;    // in 16-col units
    if (mt * 32 >= M) return;

    const int half = lane >> 4;
    const int mrow = lane & 15;
    const int Kt   = K >> 5;

    const unsigned short* pA0 = A + (size_t)(mt * 32 + mrow) * K + half * 16;
    const unsigned short* pA1 = pA0 + (size_t)16 * K;
    const unsigned short* pB0 = Bp + (size_t)(nt0 + 0) * Kt * 512 + lane * 16;
    const unsigned short* pB1 = Bp + (size_t)(nt0 + 1) * Kt * 512 + lane * 16;
    const unsigned short* pB2 = Bp + (size_t)(nt0 + 2) * Kt * 512 + lane * 16;
    const unsigned short* pB3 = Bp + (size_t)(nt0 + 3) * Kt * 512 + lane * 16;

    v8f acc[8] = {};
    BF16Frag aE[2], bE[4], aO[2], bO[4];

    auto ldA = [&](BF16Frag (&a)[2], int off) {
        a[0].us[0] = *(const v8us*)(pA0 + off);
        a[0].us[1] = *(const v8us*)(pA0 + off + 8);
        a[1].us[0] = *(const v8us*)(pA1 + off);
        a[1].us[1] = *(const v8us*)(pA1 + off + 8);
    };
    auto ldB = [&](BF16Frag (&bf)[4], int off) {
        bf[0].us[0] = *(const v8us*)(pB0 + off);
        bf[0].us[1] = *(const v8us*)(pB0 + off + 8);
        bf[1].us[0] = *(const v8us*)(pB1 + off);
        bf[1].us[1] = *(const v8us*)(pB1 + off + 8);
        bf[2].us[0] = *(const v8us*)(pB2 + off);
        bf[2].us[1] = *(const v8us*)(pB2 + off + 8);
        bf[3].us[0] = *(const v8us*)(pB3 + off);
        bf[3].us[1] = *(const v8us*)(pB3 + off + 8);
    };
    auto mma = [&](BF16Frag (&a)[2], BF16Frag (&bf)[4]) {
#pragma unroll
        for (int j = 0; j < 4; ++j) {
            acc[j]     = wmma_bf16(a[0], bf[j], acc[j]);
            acc[4 + j] = wmma_bf16(a[1], bf[j], acc[4 + j]);
        }
    };

    ldA(aE, 0);
    ldB(bE, 0);
    for (int kt = 0; kt < Kt; kt += 2) {
        // odd step's loads in flight while even step's WMMAs run
        ldA(aO, 32);
        ldB(bO, 512);
        __builtin_prefetch(pA0 + 256, 0, 1);   // global_prefetch_b8 (~4 steps ahead)
        mma(aE, bE);
        pA0 += 64;  pA1 += 64;                 // advance to kt+2
        pB0 += 1024; pB1 += 1024; pB2 += 1024; pB3 += 1024;
        if (kt + 2 < Kt) {                     // even step's loads for kt+2
            ldA(aE, 0);
            ldB(bE, 0);
        }
        mma(aO, bO);
    }

#pragma unroll
    for (int mi = 0; mi < 2; ++mi) {
#pragma unroll
        for (int j = 0; j < 4; ++j) {
            int col = (nt0 + j) * 16 + mrow;
#pragma unroll
            for (int r = 0; r < 8; ++r) {
                int row = mt * 32 + mi * 16 + r + half * 8;
                C[(size_t)row * N + col] = acc[mi * 4 + j][r];
            }
        }
    }
}

// ---------------------------------------------------------------------------
// 4) RoPE (+ bf16 convert). X,Y layout: (NTOK, heads, 128). Pair (i, i+64).
// ---------------------------------------------------------------------------
__global__ void rope_kernel(const float* __restrict__ X,
                            unsigned short* __restrict__ Y, int heads) {
    int tid   = blockIdx.x * blockDim.x + threadIdx.x;
    int total = NTOK * heads * 64;
    if (tid >= total) return;
    int i   = tid & 63;
    int hh  = (tid >> 6) % heads;
    int tok = tid / (64 * heads);
    int s   = tok & (SLEN - 1);
    float inv_freq = powf(10000.0f, -(float)i / 64.0f);
    float ang = (float)s * inv_freq;
    float c = cosf(ang), sn = sinf(ang);
    size_t base = ((size_t)tok * heads + hh) * HDIM;
    float x1 = X[base + i], x2 = X[base + i + 64];
    Y[base + i]      = f2bf(x1 * c - x2 * sn);
    Y[base + i + 64] = f2bf(x2 * c + x1 * sn);
}

// ---------------------------------------------------------------------------
// 5) Flash attention: one wave per (b, h, 16-query tile). 32 keys per step.
//    V^T staged to LDS first (stores drain under later work), then all 8 K^T
//    fragments loaded before the 8 score WMMAs, softmax, P->LDS, 8 PV WMMAs.
//    All base pointers advance by constants per key block.
// ---------------------------------------------------------------------------
__global__ __launch_bounds__(128)
void attn_kernel(const unsigned short* __restrict__ qb,
                 const unsigned short* __restrict__ kxb,
                 const unsigned short* __restrict__ vxb,
                 unsigned short* __restrict__ ob) {
    __shared__ __align__(16) unsigned short Vt[4][HDIM * 32]; // V^T tile per wave
    __shared__ __align__(16) unsigned short Ps[4][16 * 32];   // P tile per wave

    const int lane = threadIdx.x & 31;
    const int w    = threadIdx.x >> 5;
    const int wid  = blockIdx.x * 4 + w;
    const int qt   = wid & 127;          // query tile (S/16 = 128)
    const int h    = (wid >> 7) & 15;    // head
    const int b    = wid >> 11;          // batch
    const int kvh  = h >> 2;             // GQA group of 4
    unsigned short* vt = Vt[w];
    unsigned short* ps = Ps[w];

    const int half = lane >> 4;
    const int nn   = lane & 15;

    // Load Q fragments (A layout: lane = row, two 16B chunks per 32-K step)
    BF16Frag qf[4];
    const unsigned short* qrow =
        qb + ((size_t)((b * SLEN + qt * 16 + nn) * NHEAD + h)) * HDIM + half * 16;
#pragma unroll
    for (int c = 0; c < 4; ++c) {
        qf[c].us[0] = *(const v8us*)(qrow + c * 32);
        qf[c].us[1] = *(const v8us*)(qrow + c * 32 + 8);
    }

    v8f o[8] = {};
    float mrow[8], lrow[8];
#pragma unroll
    for (int r = 0; r < 8; ++r) { mrow[r] = -1e30f; lrow[r] = 0.0f; }

    const float scale = 0.08838834764831845f;   // 1/sqrt(128)
    const int kend = (qt + 1) * 16;             // causal bound for this tile

    // Base pointers; advance by 32 keys (32*NKV*HDIM elements) per block.
    const unsigned short* krow0 =
        kxb + ((size_t)((b * SLEN + nn) * NKV + kvh)) * HDIM + half * 16;
    const unsigned short* krow1 = krow0 + (size_t)16 * NKV * HDIM;
    const unsigned short* vrow =
        vxb + ((size_t)((b * SLEN + half) * NKV + kvh)) * HDIM + nn * 8;
    const size_t kstep = (size_t)32 * NKV * HDIM;

    for (int kb0 = 0; kb0 < kend;
         kb0 += 32, krow0 += kstep, krow1 += kstep, vrow += kstep) {
        // ---- stage V^T into LDS early: Vt[hd][key] for 32 keys ----
        // lane covers hd = nn*8..nn*8+7, keys = kb0 + 2t + half (t = 0..15)
#pragma unroll
        for (int t = 0; t < 16; ++t) {
            v8us gv = *(const v8us*)(vrow + (size_t)t * 2 * NKV * HDIM);
#pragma unroll
            for (int j = 0; j < 8; ++j)
                vt[(nn * 8 + j) * 32 + 2 * t + half] = gv[j];
        }

        // ---- load ALL K^T fragments, then run the 8 score WMMAs ----
        BF16Frag kf0[4], kf1[4];
#pragma unroll
        for (int c = 0; c < 4; ++c) {
            kf0[c].us[0] = *(const v8us*)(krow0 + c * 32);
            kf0[c].us[1] = *(const v8us*)(krow0 + c * 32 + 8);
            kf1[c].us[0] = *(const v8us*)(krow1 + c * 32);
            kf1[c].us[1] = *(const v8us*)(krow1 + c * 32 + 8);
        }
        v8f s0 = {}, s1 = {};
#pragma unroll
        for (int c = 0; c < 4; ++c) {
            s0 = wmma_bf16(qf[c], kf0[c], s0);
            s1 = wmma_bf16(qf[c], kf1[c], s1);
        }

        // ---- online softmax update (row lives in a 16-lane half) ----
#pragma unroll
        for (int r = 0; r < 8; ++r) {
            int   qg = qt * 16 + r + half * 8;
            int   k0 = kb0 + nn, k1 = kb0 + 16 + nn;
            float v0 = (k0 <= qg) ? s0[r] * scale : -1e30f;
            float v1 = (k1 <= qg) ? s1[r] * scale : -1e30f;
            float mx = fmaxf(v0, v1);
            mx = fmaxf(mx, __shfl_xor(mx, 1, 32));
            mx = fmaxf(mx, __shfl_xor(mx, 2, 32));
            mx = fmaxf(mx, __shfl_xor(mx, 4, 32));
            mx = fmaxf(mx, __shfl_xor(mx, 8, 32));
            float mnew = fmaxf(mrow[r], mx);
            float al   = __expf(mrow[r] - mnew);
            float p0   = __expf(v0 - mnew);
            float p1   = __expf(v1 - mnew);
            float rs   = p0 + p1;
            rs += __shfl_xor(rs, 1, 32);
            rs += __shfl_xor(rs, 2, 32);
            rs += __shfl_xor(rs, 4, 32);
            rs += __shfl_xor(rs, 8, 32);
            lrow[r] = lrow[r] * al + rs;
            mrow[r] = mnew;
#pragma unroll
            for (int nc = 0; nc < 8; ++nc) o[nc][r] *= al;
            int m = r + half * 8;
            ps[m * 32 + nn]      = f2bf(p0);
            ps[m * 32 + 16 + nn] = f2bf(p1);
        }

        // ---- O += P * V (A frag from Ps, B frags from Vt via ds_load_b128) --
        BF16Frag pa;
        pa.us[0] = *(const v8us*)(ps + nn * 32 + half * 16);
        pa.us[1] = *(const v8us*)(ps + nn * 32 + 8 + half * 16);
#pragma unroll
        for (int nc = 0; nc < 8; ++nc) {
            const unsigned short* vtb = vt + (nc * 16 + nn) * 32 + half * 16;
            BF16Frag vb;
            vb.us[0] = *(const v8us*)(vtb);
            vb.us[1] = *(const v8us*)(vtb + 8);
            o[nc] = wmma_bf16(pa, vb, o[nc]);
        }
    }

    // ---- epilogue: O/l -> bf16, layout (b, q, h, hd) ----
#pragma unroll
    for (int r = 0; r < 8; ++r) {
        float inv = 1.0f / lrow[r];
        int   m   = qt * 16 + r + half * 8;
        unsigned short* orow =
            ob + ((size_t)((b * SLEN + m) * NHEAD + h)) * HDIM + nn;
#pragma unroll
        for (int nc = 0; nc < 8; ++nc) orow[nc * 16] = f2bf(o[nc][r] * inv);
    }
}

// ---------------------------------------------------------------------------
// Launcher
// ---------------------------------------------------------------------------
extern "C" void kernel_launch(void* const* d_in, const int* in_sizes, int n_in,
                              void* d_out, int out_size, void* d_ws, size_t ws_size,
                              hipStream_t stream) {
    (void)in_sizes; (void)n_in; (void)out_size; (void)ws_size;
    const float* hs = (const float*)d_in[0];
    /* d_in[1] = attention_mask (causal) — computed analytically instead */
    const float* Wq = (const float*)d_in[2];
    const float* Wk = (const float*)d_in[3];
    const float* Wv = (const float*)d_in[4];
    const float* Wo = (const float*)d_in[5];
    float* out = (float*)d_out;

    char* ws = (char*)d_ws;
    size_t off = 0;
    auto nxt = [&](size_t bytes) {
        size_t r = off;
        off += (bytes + 255) & ~(size_t)255;
        return r;
    };
    unsigned short* hb  = (unsigned short*)(ws + nxt((size_t)NTOK * DMODEL * 2));
    unsigned short* wqP = (unsigned short*)(ws + nxt((size_t)DMODEL * 2048 * 2));
    unsigned short* wkP = (unsigned short*)(ws + nxt((size_t)DMODEL * 512 * 2));
    unsigned short* wvP = (unsigned short*)(ws + nxt((size_t)DMODEL * 512 * 2));
    unsigned short* woP = (unsigned short*)(ws + nxt((size_t)DMODEL * 2048 * 2));
    float* qf = (float*)(ws + nxt((size_t)NTOK * 2048 * 4));
    float* kf = (float*)(ws + nxt((size_t)NTOK * 512 * 4));
    float* vf = (float*)(ws + nxt((size_t)NTOK * 512 * 4));
    unsigned short* qbf = (unsigned short*)(ws + nxt((size_t)NTOK * 2048 * 2));
    unsigned short* kbf = (unsigned short*)(ws + nxt((size_t)NTOK * 512 * 2));
    unsigned short* vbf = (unsigned short*)(ws + nxt((size_t)NTOK * 512 * 2));
    unsigned short* ab  = (unsigned short*)(ws + nxt((size_t)NTOK * 2048 * 2));

    // 1) hidden -> bf16
    {
        size_t n = (size_t)NTOK * DMODEL;
        f2bf_kernel<<<(unsigned)((n + 255) / 256), 256, 0, stream>>>(hs, hb, n);
    }
    // 2) pack weights into B-fragment-major bf16
    packw_kernel<<<(64 * 128 * 32 + 255) / 256, 256, 0, stream>>>(Wq, wqP, 2048, 2048);
    packw_kernel<<<(64 * 32 * 32 + 255) / 256, 256, 0, stream>>>(Wk, wkP, 2048, 512);
    packw_kernel<<<(64 * 32 * 32 + 255) / 256, 256, 0, stream>>>(Wv, wvP, 2048, 512);
    packw_kernel<<<(64 * 128 * 32 + 255) / 256, 256, 0, stream>>>(Wo, woP, 2048, 2048);
    // 3) QKV projections (WMMA GEMMs): waves = (M/32)*(N/64), 8 waves/block
    gemm_bf16_kernel<<<512, 256, 0, stream>>>(hb, wqP, qf, NTOK, 2048, 2048);
    gemm_bf16_kernel<<<128, 256, 0, stream>>>(hb, wkP, kf, NTOK, 512, 2048);
    gemm_bf16_kernel<<<128, 256, 0, stream>>>(hb, wvP, vf, NTOK, 512, 2048);
    // 4) RoPE + bf16 conversion
    rope_kernel<<<(NTOK * 16 * 64 + 255) / 256, 256, 0, stream>>>(qf, qbf, 16);
    rope_kernel<<<(NTOK * 4 * 64 + 255) / 256, 256, 0, stream>>>(kf, kbf, 4);
    {
        size_t n = (size_t)NTOK * 512;
        f2bf_kernel<<<(unsigned)((n + 255) / 256), 256, 0, stream>>>(vf, vbf, n);
    }
    // 5) Flash attention (4096 waves, 4 waves/block)
    attn_kernel<<<1024, 128, 0, stream>>>(qbf, kbf, vbf, ab);
    // 6) Output projection -> f32 d_out
    gemm_bf16_kernel<<<512, 256, 0, stream>>>(ab, woP, out, NTOK, 2048, 2048);
}
